// Net_88433376624809
// MI455X (gfx1250) — matrix operand order repaired and checked by reading, not compiled, not run on hardware
//
#include <hip/hip_runtime.h>
#include <hip/hip_bf16.h>
#include <math.h>

#define B_   512
#define S_   256
#define NC_  5000
#define HID_ 512
#define MLP_ 512
#define FEAT_ 22
#define INP_ 32   // IN=24 zero-padded to 32 so U@x is one WMMA K-chunk

typedef _Float16 v16h __attribute__((ext_vector_type(16)));
typedef _Float16 h8   __attribute__((ext_vector_type(8)));
typedef float    v8f  __attribute__((ext_vector_type(8)));

#define EPSN 1e-15f
#define ATCLIP (1.0f - 1e-5f)

__device__ __forceinline__ float clampn2(float sq) {      // clipped norm from sum of squares
  return fmaxf(sqrtf(fmaxf(sq, 0.f)), EPSN);
}
__device__ __forceinline__ float artanhc(float x) {
  x = fminf(fmaxf(x, -ATCLIP), ATCLIP);
  return 0.5f * logf((1.f + x) / (1.f - x));
}
__device__ __forceinline__ float sigmoidf_(float x) { return 1.f / (1.f + expf(-x)); }

__device__ __forceinline__ v8f v8f_zero() {
  v8f z;
#pragma unroll
  for (int j = 0; j < 8; ++j) z[j] = 0.f;
  return z;
}

// Optimization barrier that PRESERVES address-space provenance: we launder a zero
// *offset* (not the pointer), so the base pointer still infers as global(1) and the
// loads stay global_load_b128 (not flat_load, which ties up the LDS path + DScnt).
// The volatile asm executes inside the step loop, making derived addresses
// loop-variant so LLVM cannot hoist the weight loads out of the scan and spill them.
__device__ __forceinline__ unsigned long long opaque_zero() {
  unsigned long long z = 0;
  asm volatile("" : "+v"(z));
  return z;
}

__device__ __forceinline__ v8f wmma16(v16h a, v16h b, v8f c) {
  // v_wmma_f32_16x16x32_f16: (neg_a, A, neg_b, B, c_mod, C, reuse_a, reuse_b)
  return __builtin_amdgcn_wmma_f32_16x16x32_f16(false, a, false, b, (short)0, c, false, false);
}

// ---- A fragment: 16x32 f16, rows in `mat` row-major with `stride` halves.
// lanes 0-15: M=lane, K = k0+{0..7} U {16..23}; lanes 16-31: M=lane-16, K = k0+{8..15} U {24..31}
__device__ __forceinline__ v16h loadAfrag(const _Float16* mat, int stride, int k0, int lane) {
  int r = lane & 15;
  int koff = (lane < 16) ? 0 : 8;
  const _Float16* p = mat + (size_t)r * stride + k0 + koff;
  h8 lo = *(const h8*)p;
  h8 hi = *(const h8*)(p + 16);
  v16h a;
#pragma unroll
  for (int j = 0; j < 8; ++j) { a[j] = lo[j]; a[j + 8] = hi[j]; }
  return a;
}

// ---- A fragment built on the fly from fp32 rows (for h_last @ mlp_W.T)
__device__ __forceinline__ v16h loadAfrag_f32(const float* mat, int stride, int k0, int lane) {
  int r = lane & 15;
  int koff = (lane < 16) ? 0 : 8;
  const float* p = mat + (size_t)r * stride + k0 + koff;
  v16h a;
#pragma unroll
  for (int j = 0; j < 8; ++j) { a[j] = (_Float16)p[j]; a[j + 8] = (_Float16)p[j + 16]; }
  return a;
}

// ---- B fragment: 32x16 f16 from W[n][k] row-major (out = in @ W.T, so B[k][n] = W[n][k]).
// lanes = N column; lanes 0-15 hold K=k0+{0..7} U {16..23}, lanes 16-31 hold K=k0+{8..15} U {24..31}.
__device__ __forceinline__ v16h loadBfrag(const _Float16* W, int stride, int n0, int k0, int lane) {
  int n = n0 + (lane & 15);
  int k = k0 + ((lane < 16) ? 0 : 8);
  const _Float16* p = W + (size_t)n * stride + k;
  h8 lo = *(const h8*)p;
  h8 hi = *(const h8*)(p + 16);
  v16h b;
#pragma unroll
  for (int j = 0; j < 8; ++j) { b[j] = lo[j]; b[j + 8] = hi[j]; }
  return b;
}

// ---- deterministic per-row reduction across the 32 waves of a 1024-thread block.
// C-frag layout: lanes 0-15 hold rows 0..7 (vgpr j), lanes 16-31 hold rows 8..15.
__device__ __forceinline__ void rowPart(float (*part)[32][16], int ch, float v[8], int lane, int wave) {
#pragma unroll
  for (int m = 1; m < 16; m <<= 1) {
#pragma unroll
    for (int j = 0; j < 8; ++j) v[j] += __shfl_xor(v[j], m, 32);
  }
  if ((lane & 15) == 0) {
    int base = (lane < 16) ? 0 : 8;
#pragma unroll
    for (int j = 0; j < 8; ++j) part[ch][wave][base + j] = v[j];
  }
}
__device__ __forceinline__ void rowFinish(float (*part)[32][16], float* red, int tid) {
  __syncthreads();
  if (tid < 32) {
    int ch = tid >> 4, row = tid & 15;
    float s = 0.f;
#pragma unroll 8
    for (int w = 0; w < 32; ++w) s += part[ch][w][row];
    red[tid] = s;
  }
  __syncthreads();
}

// ======================================================================
// one_rnn_transform = mobius_add(mobius_add(mobius_matvec(W,A), mobius_matvec(U,x)), b)
// A lives in LDS (f16, [16][512]); an2 = per-row ||A||^2; xn2row = per-row ||x||^2.
// Returns a2 tile (v8f per wave) and per-row ||a2||^2 in na2_out.
// ======================================================================
__device__ void mobius_gate(
    const _Float16* __restrict__ Wg,   // [512][512] f16
    const _Float16* __restrict__ Ug,   // [512][32]  f16 (padded)
    float bcol, float bn2g,
    const _Float16* ldsA, const float* an2, const float* xn2row,
    v16h xfrag, float (*part)[32][16], float* red, float* na2_out,
    int lane, int tid, int wave, int n0, v8f& a2_out)
{
  const int rb = (lane < 16) ? 0 : 8;
  Wg += opaque_zero();        // loop-variant address: keep weight streaming in-loop (L2)
  Ug += opaque_zero();
  __syncthreads();                                   // ldsA ready for all waves

  // Software-pipelined K loop: issue loads for chunk kc+1 before WMMA of chunk kc
  // so the scheduler can overlap L2 latency with matrix math (partial waits).
  v8f acc = v8f_zero();
  v16h a_cur = loadAfrag(ldsA, HID_, 0, lane);
  v16h b_cur = loadBfrag(Wg, HID_, n0, 0, lane);
#pragma unroll
  for (int kc = 0; kc < 16; ++kc) {
    v16h a_nxt, b_nxt;
    if (kc < 15) {
      a_nxt = loadAfrag(ldsA, HID_, (kc + 1) * 32, lane);
      b_nxt = loadBfrag(Wg, HID_, n0, (kc + 1) * 32, lane);
    }
    acc = wmma16(a_cur, b_cur, acc);
    if (kc < 15) { a_cur = a_nxt; b_cur = b_nxt; }
  }
  v8f accx = v8f_zero();
  {
    v16h b = loadBfrag(Ug, INP_, n0, 0, lane);
    accx = wmma16(xfrag, b, accx);
  }

  // R1: ||Wh||^2 and ||Ux||^2 per row
  __syncthreads();
  { float t1[8], t2[8];
#pragma unroll
    for (int j = 0; j < 8; ++j) { t1[j] = acc[j] * acc[j]; t2[j] = accx[j] * accx[j]; }
    rowPart(part, 0, t1, lane, wave); rowPart(part, 1, t2, lane, wave);
  }
  rowFinish(part, red, tid);

  float p[8], q[8], c1s[8], c2s[8];
#pragma unroll
  for (int j = 0; j < 8; ++j) {
    int row = rb + j;
    float m2 = red[row];
    if (m2 <= 0.f) { p[j] = 0.f; c1s[j] = 0.f; }
    else { float mn = clampn2(m2); float xn = clampn2(an2[row]);
           float c = tanhf(mn / xn * artanhc(xn)); p[j] = c / mn * acc[j]; c1s[j] = c; }
    float m2x = red[16 + row];
    if (m2x <= 0.f) { q[j] = 0.f; c2s[j] = 0.f; }
    else { float mn = clampn2(m2x); float xn = clampn2(xn2row[row]);
           float c = tanhf(mn / xn * artanhc(xn)); q[j] = c / mn * accx[j]; c2s[j] = c; }
  }

  // R2: xy = <p,q> per row ; ||p||^2 = c1^2, ||q||^2 = c2^2 (analytic)
  __syncthreads();
  { float pq[8];
#pragma unroll
    for (int j = 0; j < 8; ++j) pq[j] = p[j] * q[j];
    rowPart(part, 0, pq, lane, wave);
  }
  rowFinish(part, red, tid);

  float a1[8];
#pragma unroll
  for (int j = 0; j < 8; ++j) {
    float xy = red[rb + j], x2 = c1s[j] * c1s[j], y2 = c2s[j] * c2s[j];
    float den = fmaxf(1.f + 2.f * xy + x2 * y2, EPSN);
    a1[j] = ((1.f + 2.f * xy + y2) * p[j] + (1.f - x2) * q[j]) / den;
  }

  // R3: ||a1||^2 and <a1,b> per row
  __syncthreads();
  { float aa[8], ab[8];
#pragma unroll
    for (int j = 0; j < 8; ++j) { aa[j] = a1[j] * a1[j]; ab[j] = a1[j] * bcol; }
    rowPart(part, 0, aa, lane, wave); rowPart(part, 1, ab, lane, wave);
  }
  rowFinish(part, red, tid);
  float x2r[8], xyr[8];
#pragma unroll
  for (int j = 0; j < 8; ++j) { x2r[j] = red[rb + j]; xyr[j] = red[16 + rb + j]; }

  // a2 = mobius_add(a1, b) ; R4: ||a2||^2
  __syncthreads();
  v8f a2;
  { float a22[8];
#pragma unroll
    for (int j = 0; j < 8; ++j) {
      float den = fmaxf(1.f + 2.f * xyr[j] + x2r[j] * bn2g, EPSN);
      a2[j] = ((1.f + 2.f * xyr[j] + bn2g) * a1[j] + (1.f - x2r[j]) * bcol) / den;
      a22[j] = a2[j] * a2[j];
    }
    rowPart(part, 0, a22, lane, wave);
  }
  rowFinish(part, red, tid);
  if (tid < 16) na2_out[tid] = red[tid];
  __syncthreads();
  a2_out = a2;
}

// ======================================================================
// Persistent Mobius-GRU: each workgroup owns 16 batch rows for all 256 steps.
// 32 waves: wave w computes output columns [16w,16w+16). No cross-WG sync needed.
// ======================================================================
__global__ __launch_bounds__(1024)
void mobius_gru_kernel(const _Float16* __restrict__ WhhH,   // [3*512][512]
                       const _Float16* __restrict__ WihH,   // [3*512][32]
                       const float*    __restrict__ biasH,  // [3][512]
                       const float*    __restrict__ bn2,    // [3]
                       const _Float16* __restrict__ Xh,     // [S][B][32]
                       const float*    __restrict__ xn2g,   // [S][B]
                       float*          __restrict__ Hout)   // [B][512]
{
  __shared__ __align__(16) float    hF[16 * HID_];     // fp32 master h (32 KB)
  __shared__ __align__(16) _Float16 hAB[16 * HID_];    // f16 WMMA A operand: h, then rh (16 KB)
  __shared__ float part[2][32][16];                    // deterministic reduction partials (4 KB)
  __shared__ float red[32];
  __shared__ float hn2[16], xrow[16], rhc2[16], na2[16];

  const int tid  = threadIdx.x;
  const int lane = tid & 31;
  const int wave = tid >> 5;
  const int m0   = blockIdx.x * 16;
  const int n0   = wave * 16;
  const int ncol = n0 + (lane & 15);
  const int rb   = (lane < 16) ? 0 : 8;

  for (int i = tid; i < 16 * HID_; i += 1024) hF[i] = 0.f;
  if (tid < 16) hn2[tid] = 0.f;
  __syncthreads();

  // reference gate order: split rows -> [r | h~ | z]
  const float br = biasH[0 * HID_ + ncol];
  const float bh = biasH[1 * HID_ + ncol];
  const float bz = biasH[2 * HID_ + ncol];
  const float bn2r = bn2[0], bn2h = bn2[1], bn2z = bn2[2];

  for (int s = 0; s < S_; ++s) {
    for (int i = tid; i < 16 * HID_; i += 1024) hAB[i] = (_Float16)hF[i];
    if (tid < 16) xrow[tid] = xn2g[s * B_ + m0 + tid];
    __syncthreads();

    v16h xfrag = loadAfrag(Xh + ((size_t)s * B_ + m0) * INP_, INP_, 0, lane);

    // ---- z gate ----
    float zg[8];
    {
      v8f a2;
      mobius_gate(WhhH + 2 * HID_ * HID_, WihH + 2 * HID_ * INP_, bz, bn2z,
                  hAB, hn2, xrow, xfrag, part, red, na2, lane, tid, wave, n0, a2);
#pragma unroll
      for (int j = 0; j < 8; ++j) {
        float n = clampn2(na2[rb + j]);
        zg[j] = sigmoidf_(artanhc(n) / n * a2[j]);      // sigmoid(logmap0(a2))
      }
    }
    // ---- r gate, then rh = mobius_pointwise_mul(r, h) (overwrites hAB) ----
    {
      v8f a2;
      mobius_gate(WhhH, WihH, br, bn2r,
                  hAB, hn2, xrow, xfrag, part, red, na2, lane, tid, wave, n0, a2);
      float w[8];
#pragma unroll
      for (int j = 0; j < 8; ++j) {
        float n = clampn2(na2[rb + j]);
        float rg = sigmoidf_(artanhc(n) / n * a2[j]);
        w[j] = rg * hF[(rb + j) * HID_ + ncol];
      }
      __syncthreads();
      { float w2[8];
#pragma unroll
        for (int j = 0; j < 8; ++j) w2[j] = w[j] * w[j];
        rowPart(part, 0, w2, lane, wave);
      }
      rowFinish(part, red, tid);
#pragma unroll
      for (int j = 0; j < 8; ++j) {
        int row = rb + j;
        float m2 = red[row], rhv, c;
        if (m2 <= 0.f) { rhv = 0.f; c = 0.f; }
        else {
          float mn = clampn2(m2), xn = clampn2(hn2[row]);
          c = tanhf(mn / xn * artanhc(xn));
          rhv = c / mn * w[j];
        }
        hAB[row * HID_ + ncol] = (_Float16)rhv;
        if (wave == 0) rhc2[row] = c * c;               // analytic ||rh||^2
      }
      __syncthreads();
    }
    // ---- h_tilde = one_rnn_transform(W_hh, rh, W_ih, x, b_h) ----
    v8f ht;
    mobius_gate(WhhH + 1 * HID_ * HID_, WihH + 1 * HID_ * INP_, bh, bn2h,
                hAB, rhc2, xrow, xfrag, part, red, na2, lane, tid, wave, n0, ht);
    float htn2[8], hval[8];
#pragma unroll
    for (int j = 0; j < 8; ++j) { htn2[j] = na2[rb + j]; hval[j] = hF[(rb + j) * HID_ + ncol]; }

    // dh = mobius_add(-h, h_tilde): xy = <-h, ht>
    __syncthreads();
    { float pr[8];
#pragma unroll
      for (int j = 0; j < 8; ++j) pr[j] = -hval[j] * ht[j];
      rowPart(part, 0, pr, lane, wave);
    }
    rowFinish(part, red, tid);
    float dh[8];
#pragma unroll
    for (int j = 0; j < 8; ++j) {
      float xy = red[rb + j], x2 = hn2[rb + j], y2 = htn2[j];
      float den = fmaxf(1.f + 2.f * xy + x2 * y2, EPSN);
      dh[j] = ((1.f + 2.f * xy + y2) * (-hval[j]) + (1.f - x2) * ht[j]) / den;
    }
    // ||dh||^2 and ||z*dh||^2 (fused)
    __syncthreads();
    { float d2[8], zd2[8];
#pragma unroll
      for (int j = 0; j < 8; ++j) { float zd = zg[j] * dh[j]; d2[j] = dh[j] * dh[j]; zd2[j] = zd * zd; }
      rowPart(part, 0, d2, lane, wave); rowPart(part, 1, zd2, lane, wave);
    }
    rowFinish(part, red, tid);
    float pv[8], c2v[8];
#pragma unroll
    for (int j = 0; j < 8; ++j) {
      float dd = red[rb + j], zz = red[16 + rb + j];
      if (zz <= 0.f) { pv[j] = 0.f; c2v[j] = 0.f; }
      else {
        float mn = clampn2(zz), xn = clampn2(dd);
        float c = tanhf(mn / xn * artanhc(xn));
        pv[j] = c / mn * zg[j] * dh[j]; c2v[j] = c * c;
      }
    }
    // hn = mobius_add(h, p): xy = <h,p>
    __syncthreads();
    { float hp[8];
#pragma unroll
      for (int j = 0; j < 8; ++j) hp[j] = hval[j] * pv[j];
      rowPart(part, 0, hp, lane, wave);
    }
    rowFinish(part, red, tid);
    float hnv[8];
#pragma unroll
    for (int j = 0; j < 8; ++j) {
      float xy = red[rb + j], x2 = hn2[rb + j], y2 = c2v[j];
      float den = fmaxf(1.f + 2.f * xy + x2 * y2, EPSN);
      hnv[j] = ((1.f + 2.f * xy + y2) * hval[j] + (1.f - x2) * pv[j]) / den;
    }
    // write back h and refresh ||h||^2
    __syncthreads();
    { float h2[8];
#pragma unroll
      for (int j = 0; j < 8; ++j) { hF[(rb + j) * HID_ + ncol] = hnv[j]; h2[j] = hnv[j] * hnv[j]; }
      rowPart(part, 0, h2, lane, wave);
    }
    rowFinish(part, red, tid);
    if (tid < 16) hn2[tid] = red[tid];
    __syncthreads();
  }

  for (int i = tid; i < 16 * HID_; i += 1024) Hout[(size_t)m0 * HID_ + i] = hF[i];
}

// ======================================================================
// Head: m = tanh(H @ mlp_W.T + b) -> f16
// ======================================================================
__global__ __launch_bounds__(256)
void head_mlp_kernel(const float* __restrict__ H, const _Float16* __restrict__ mlpH,
                     const float* __restrict__ mlp_b, _Float16* __restrict__ Mh)
{
  int tid = threadIdx.x, lane = tid & 31, wave = tid >> 5;
  int gid = blockIdx.x * 8 + wave;          // 1024 tiles: 32 M x 32 N
  int m0 = (gid >> 5) * 16, n0 = (gid & 31) * 16;
  v8f acc = v8f_zero();
  for (int kc = 0; kc < 16; ++kc) {
    v16h a = loadAfrag_f32(H + (size_t)m0 * HID_, HID_, kc * 32, lane);
    v16h b = loadBfrag(mlpH, HID_, n0, kc * 32, lane);
    acc = wmma16(a, b, acc);
  }
  int n = n0 + (lane & 15);
  float bc = mlp_b[n];
  int rbm = (lane < 16) ? 0 : 8;
#pragma unroll
  for (int j = 0; j < 8; ++j)
    Mh[(size_t)(m0 + rbm + j) * MLP_ + n] = (_Float16)tanhf(acc[j] + bc);
}

// event_out = m @ ev_W.T + ev_b   (512 x 5000 x 512)
__global__ __launch_bounds__(256)
void head_event_kernel(const _Float16* __restrict__ Mh, const _Float16* __restrict__ evH,
                       const float* __restrict__ ev_b, float* __restrict__ out)
{
  int tid = threadIdx.x, lane = tid & 31, wave = tid >> 5;
  int gid = blockIdx.x * 8 + wave;          // 32 * 313 = 10016 tiles exactly
  int m0 = (gid / 313) * 16, n0 = (gid % 313) * 16;
  int n = n0 + (lane & 15);
  bool valid = (n < NC_);
  const _Float16* bp = evH + (size_t)(valid ? n : 0) * MLP_;
  v8f acc = v8f_zero();
  for (int kc = 0; kc < 16; ++kc) {
    v16h a = loadAfrag(Mh + (size_t)m0 * MLP_, MLP_, kc * 32, lane);
    int k = kc * 32 + ((lane < 16) ? 0 : 8);
    const _Float16* p = bp + k;
    h8 lo = *(const h8*)p, hi = *(const h8*)(p + 16);
    v16h b;
#pragma unroll
    for (int j = 0; j < 8; ++j) { b[j] = lo[j]; b[j + 8] = hi[j]; }
    if (!valid) {
#pragma unroll
      for (int j = 0; j < 16; ++j) b[j] = (_Float16)0.f;
    }
    acc = wmma16(a, b, acc);
  }
  float bc = valid ? ev_b[n] : 0.f;
  int rbm = (lane < 16) ? 0 : 8;
#pragma unroll
  for (int j = 0; j < 8; ++j)
    if (valid) out[(size_t)(m0 + rbm + j) * NC_ + n] = acc[j] + bc;
}

// time_out = m @ t_W.T + t_b  (512 x 1)
__global__ __launch_bounds__(256)
void head_time_kernel(const _Float16* __restrict__ Mh, const float* __restrict__ tW,
                      const float* __restrict__ tb, float* __restrict__ out)
{
  __shared__ float sm[256];
  int b = blockIdx.x, tid = threadIdx.x;
  float acc = 0.f;
  for (int k = tid; k < MLP_; k += 256) acc += (float)Mh[(size_t)b * MLP_ + k] * tW[k];
  sm[tid] = acc; __syncthreads();
  for (int st = 128; st > 0; st >>= 1) { if (tid < st) sm[tid] += sm[tid + st]; __syncthreads(); }
  if (tid == 0) out[b] = sm[0] + tb[0];
}

// ======================================================================
// Prologue kernels
// ======================================================================
__global__ void prep_weights_kernel(const float* __restrict__ Whh, const float* __restrict__ Wih,
                                    const float* __restrict__ mlpW, const float* __restrict__ evW,
                                    const float* __restrict__ bias,
                                    _Float16* __restrict__ WhhH, _Float16* __restrict__ WihH,
                                    _Float16* __restrict__ mlpH, _Float16* __restrict__ evH,
                                    float* __restrict__ bn2)
{
  int stride = gridDim.x * blockDim.x;
  int t0 = blockIdx.x * blockDim.x + threadIdx.x;
  for (int i = t0; i < 3 * HID_ * HID_; i += stride) WhhH[i] = (_Float16)Whh[i];
  for (int i = t0; i < 3 * HID_ * INP_; i += stride) {
    int k = i & 31, row = i >> 5;
    WihH[i] = (k < 24) ? (_Float16)Wih[row * 24 + k] : (_Float16)0.f;
  }
  for (int i = t0; i < MLP_ * HID_; i += stride) mlpH[i] = (_Float16)mlpW[i];
  for (int i = t0; i < NC_ * MLP_; i += stride) evH[i] = (_Float16)evW[i];
  if (t0 < 3) {
    float s = 0.f;
    for (int k = 0; k < HID_; ++k) { float v = bias[t0 * HID_ + k]; s += v * v; }
    bn2[t0] = s;
  }
}

// Gexp = expmap0(gcn_out) / 10
__global__ void prep_gexp_kernel(const float* __restrict__ g, float* __restrict__ Gexp)
{
  int i = blockIdx.x * blockDim.x + threadIdx.x;
  if (i >= NC_) return;
  float v[FEAT_]; float ss = 0.f;
#pragma unroll
  for (int j = 0; j < FEAT_; ++j) { v[j] = g[i * FEAT_ + j]; ss += v[j] * v[j]; }
  float n = clampn2(ss);
  float f = tanhf(n) / n * 0.1f;
#pragma unroll
  for (int j = 0; j < FEAT_; ++j) Gexp[i * FEAT_ + j] = f * v[j];
}

// tfac[b] = tanh(||input_time[b,:]||)/||.|| / 10   (expmap0 over the seq axis)
__global__ __launch_bounds__(256)
void prep_tfac_kernel(const float* __restrict__ it, float* __restrict__ tfac)
{
  __shared__ float sm[256];
  int b = blockIdx.x, tid = threadIdx.x;
  float v = it[b * S_ + tid];
  sm[tid] = v * v; __syncthreads();
  for (int st = 128; st > 0; st >>= 1) { if (tid < st) sm[tid] += sm[tid + st]; __syncthreads(); }
  if (tid == 0) { float n = clampn2(sm[0]); tfac[b] = tanhf(n) / n * 0.1f; }
}

// Xh[s][b][0..31] (f16, padded) and xn2[s][b] = ||x||^2 (fp32)
__global__ void prep_x_kernel(const int* __restrict__ ev, const float* __restrict__ it,
                              const float* __restrict__ Gexp, const float* __restrict__ emb,
                              const float* __restrict__ tfac,
                              _Float16* __restrict__ Xh, float* __restrict__ xn2)
{
  int i = blockIdx.x * blockDim.x + threadIdx.x;   // i = s*512 + b
  int s = i >> 9, b = i & 511;
  int cls = ev[b * S_ + s];
  _Float16* o = Xh + (size_t)i * INP_;
  float ss = 0.f;
#pragma unroll
  for (int j = 0; j < FEAT_; ++j) { float v = Gexp[cls * FEAT_ + j]; o[j] = (_Float16)v; ss += v * v; }
  float nd = emb[cls]; o[22] = (_Float16)nd; ss += nd * nd;
  float tv = it[b * S_ + s] * tfac[b]; o[23] = (_Float16)tv; ss += tv * tv;
#pragma unroll
  for (int j = 24; j < 32; ++j) o[j] = (_Float16)0.f;
  xn2[i] = ss;
}

// ======================================================================
extern "C" void kernel_launch(void* const* d_in, const int* in_sizes, int n_in,
                              void* d_out, int out_size, void* d_ws, size_t ws_size,
                              hipStream_t stream)
{
  const float* input_time = (const float*)d_in[0];
  const int*   event_in   = (const int*)d_in[1];
  const float* gcn_out    = (const float*)d_in[2];
  const float* emb_table  = (const float*)d_in[3];
  const float* W_ih       = (const float*)d_in[4];
  const float* W_hh       = (const float*)d_in[5];
  const float* bias_h     = (const float*)d_in[6];
  const float* mlp_W      = (const float*)d_in[7];
  const float* mlp_b      = (const float*)d_in[8];
  const float* ev_W       = (const float*)d_in[9];
  const float* ev_b       = (const float*)d_in[10];
  const float* t_W        = (const float*)d_in[11];
  const float* t_b        = (const float*)d_in[12];
  (void)in_sizes; (void)n_in; (void)out_size; (void)ws_size;

  char* ws = (char*)d_ws;
  size_t off = 0;
  auto take = [&](size_t bytes) -> char* {
    char* p = ws + off;
    off = (off + bytes + 255) & ~(size_t)255;
    return p;
  };
  _Float16* WhhH = (_Float16*)take((size_t)3 * HID_ * HID_ * 2);
  _Float16* WihH = (_Float16*)take((size_t)3 * HID_ * INP_ * 2);
  _Float16* mlpH = (_Float16*)take((size_t)MLP_ * HID_ * 2);
  _Float16* evH  = (_Float16*)take((size_t)NC_ * MLP_ * 2);
  float*    bn2  = (float*)take(16);
  float*    Gexp = (float*)take((size_t)NC_ * FEAT_ * 4);
  float*    tfac = (float*)take((size_t)B_ * 4);
  _Float16* Xh   = (_Float16*)take((size_t)S_ * B_ * INP_ * 2);
  float*    xn2  = (float*)take((size_t)S_ * B_ * 4);
  float*    Hbuf = (float*)take((size_t)B_ * HID_ * 4);
  _Float16* Mh   = (_Float16*)take((size_t)B_ * MLP_ * 2);
  // total ~18.3 MB of d_ws

  prep_weights_kernel<<<512, 256, 0, stream>>>(W_hh, W_ih, mlp_W, ev_W, bias_h,
                                               WhhH, WihH, mlpH, evH, bn2);
  prep_gexp_kernel<<<(NC_ + 255) / 256, 256, 0, stream>>>(gcn_out, Gexp);
  prep_tfac_kernel<<<B_, 256, 0, stream>>>(input_time, tfac);
  prep_x_kernel<<<(S_ * B_) / 256, 256, 0, stream>>>(event_in, input_time, Gexp, emb_table,
                                                     tfac, Xh, xn2);
  mobius_gru_kernel<<<B_ / 16, 1024, 0, stream>>>(WhhH, WihH, bias_h, bn2, Xh, xn2, Hbuf);
  head_mlp_kernel<<<128, 256, 0, stream>>>(Hbuf, mlpH, mlp_b, Mh);
  head_event_kernel<<<1252, 256, 0, stream>>>(Mh, evH, ev_b, (float*)d_out);
  head_time_kernel<<<B_, 256, 0, stream>>>(Mh, t_W, t_b, (float*)d_out + (size_t)B_ * NC_);
}
